// DeformableTransformer_34196529610795
// MI455X (gfx1250) — matrix operand order repaired and checked by reading, not compiled, not run on hardware
//
#include <hip/hip_runtime.h>

// ---- problem constants (match reference) ----
#define NLAYER 6
#define DMODEL 256
#define NHEAD 8
#define DHEAD 32
#define LQn 1800
#define LVL 4
#define PTn 8          // NP + TW
#define NPn 4
#define TWn 4
#define TFR 6
#define DFF 1024
#define LENn 110160    // sum over levels of 6*W_l

typedef __attribute__((ext_vector_type(16))) _Float16 v16h;
typedef __attribute__((ext_vector_type(8)))  float    v8f;

// ---- wave32 WMMA fragment index helpers (CDNA5 ISA 7.12.2 layouts) ----
// 16-bit A 16x32, contiguous K order chosen here (ISA table order differs): lanes 0-15 row=lane K:{0..15}; lanes 16-31 row=lane-16 K:{16..31}
// => per lane the needed K values are one contiguous run of 16 floats: [base0,base0+16)
__device__ __forceinline__ int a_base0(int lane) { return (lane < 16) ? 0 : 16; }
// 16-bit B 32x16, same contiguous K order as A here (ISA table order differs): vgpr j pair: lanes 0-15 K=2j,2j+1 ; lanes 16-31 K=16+2j ; col=lane&15
__device__ __forceinline__ int b_kk(int lane, int j) { return (lane < 16 ? 0 : 16) + 2 * j; }
// C/D 16x16 f32: vgpr r -> row r (lanes 0-15) / row r+8 (lanes 16-31); col = lane&15

__device__ __forceinline__ void cvt8(v16h& dst, int o, float4 a, float4 b) {
  dst[o + 0] = (_Float16)a.x; dst[o + 1] = (_Float16)a.y;
  dst[o + 2] = (_Float16)a.z; dst[o + 3] = (_Float16)a.w;
  dst[o + 4] = (_Float16)b.x; dst[o + 5] = (_Float16)b.y;
  dst[o + 6] = (_Float16)b.z; dst[o + 7] = (_Float16)b.w;
}

// =======================================================================
// Weight repack: f32 (K,N) row-major -> f16 in WMMA B-fragment order:
// Wp[(((nt*kchunks)+kc)*32 + lane)*16 + e] = W[kc*32 + (lane<16?0:16) + e][nt*16 + (lane&15)]
// so a lane's whole fragment is one aligned 32B v16h load.
// =======================================================================
__global__ void pack_w_f16(const float* __restrict__ W, _Float16* __restrict__ Wp,
                           int K, int N) {
  const int i = blockIdx.x * blockDim.x + threadIdx.x;
  if (i >= K * N) return;
  const int kchunks = K >> 5;
  const int e = i & 15;
  const int lane = (i >> 4) & 31;
  const int kc = (i >> 9) % kchunks;
  const int nt = i / (kchunks << 9);
  const int col = nt * 16 + (lane & 15);
  const int krow = (kc << 5) + ((lane < 16) ? 0 : 16) + e;
  Wp[i] = (_Float16)W[(size_t)krow * N + col];
}

// =======================================================================
// GEMM: C[M,N] = A[M,K](f32) @ Wp[K,N](packed f16) + bias ; f32 accum.
// One wave computes 64x32 (4 M-subtiles x 2 N-subtiles). Rows clamped
// (never stored OOB) so all loads are unconditional b128 clauses.
// =======================================================================
#define GMT 4
#define GNT 2
__global__ __launch_bounds__(32, 1) void gemm_wmma(
    const float* __restrict__ A, const _Float16* __restrict__ Wp,
    const float* __restrict__ bias, float* __restrict__ Cf,
    _Float16* __restrict__ Ch, int M, int N, int K, int relu) {
  const int lane = threadIdx.x;
  const int m0 = blockIdx.x * (16 * GMT);
  const int n0t = blockIdx.y * GNT;           // N-subtile index base
  const int arow = lane & 15;
  const int kchunks = K >> 5;
  const int b0 = a_base0(lane);

  int rowc[GMT];
#pragma unroll
  for (int mt = 0; mt < GMT; ++mt) {
    int r = m0 + mt * 16 + arow;
    rowc[mt] = (r < M) ? r : (M - 1);
  }

  v8f acc[GMT][GNT];
#pragma unroll
  for (int mt = 0; mt < GMT; ++mt)
#pragma unroll
    for (int nt = 0; nt < GNT; ++nt) { v8f z = {}; acc[mt][nt] = z; }

  for (int kc = 0; kc < kchunks; ++kc) {
    const int k0 = kc << 5;
    // B fragments: one aligned v16h load each (2 x b128)
    v16h bfr[GNT];
#pragma unroll
    for (int nt = 0; nt < GNT; ++nt)
      bfr[nt] = *(const v16h*)(Wp + ((((size_t)(n0t + nt)) * kchunks + kc) * 32 + lane) * 16);
    // A fragments one at a time (short live range, no spills)
#pragma unroll
    for (int mt = 0; mt < GMT; ++mt) {
      const float* ap = A + (size_t)rowc[mt] * K + k0;
      __builtin_prefetch(ap + 64, 0, 1);   // stream-ahead on A (global_prefetch)
      const float4 a0 = *(const float4*)(ap + b0);
      const float4 a1 = *(const float4*)(ap + b0 + 4);
      const float4 a2 = *(const float4*)(ap + b0 + 8);
      const float4 a3 = *(const float4*)(ap + b0 + 12);
      v16h afr;
      cvt8(afr, 0, a0, a1);
      cvt8(afr, 8, a2, a3);
#pragma unroll
      for (int nt = 0; nt < GNT; ++nt)
        acc[mt][nt] = __builtin_amdgcn_wmma_f32_16x16x32_f16(
            false, afr, false, bfr[nt], (short)0, acc[mt][nt], false, false);
    }
  }

#pragma unroll
  for (int mt = 0; mt < GMT; ++mt)
#pragma unroll
    for (int nt = 0; nt < GNT; ++nt) {
      const int col = (n0t + nt) * 16 + (lane & 15);
      const float bv = bias ? bias[col] : 0.f;
#pragma unroll
      for (int r = 0; r < 8; ++r) {
        const int row = m0 + mt * 16 + r + (lane < 16 ? 0 : 8);
        if (row < M) {
          float v = acc[mt][nt][r] + bv;
          if (relu) v = fmaxf(v, 0.f);
          if (Ch) Ch[(size_t)row * N + col] = (_Float16)v;
          else    Cf[(size_t)row * N + col] = v;
        }
      }
    }
}

// =======================================================================
// Flash self-attention: one wave per (16-query tile, head). DH = 32.
// QK^T and P*V via WMMA; P transposed C->A layout through LDS.
// Indices clamped; invalid key columns masked to -inf pre-softmax.
// =======================================================================
__global__ __launch_bounds__(32, 1) void flash_attn(
    const float* __restrict__ Qp, const float* __restrict__ Kp,
    const float* __restrict__ Vp, float* __restrict__ O, int Lq) {
  __shared__ float pbuf[16][33];
  const int lane = threadIdx.x;
  const int h = blockIdx.y;
  const int q0 = blockIdx.x * 16;
  const float scale = 0.17677669529663687f;  // 1/sqrt(32)
  const int b0 = a_base0(lane);

  // Q fragment (A layout), scale folded in; clamped row (never stored OOB)
  v16h qa;
  {
    int qrow = q0 + (lane & 15);
    if (qrow >= Lq) qrow = Lq - 1;
    const float* qp = Qp + (size_t)qrow * DMODEL + h * DHEAD;
    float4 a0 = *(const float4*)(qp + b0);
    float4 a1 = *(const float4*)(qp + b0 + 4);
    float4 a2 = *(const float4*)(qp + b0 + 8);
    float4 a3 = *(const float4*)(qp + b0 + 12);
    a0.x *= scale; a0.y *= scale; a0.z *= scale; a0.w *= scale;
    a1.x *= scale; a1.y *= scale; a1.z *= scale; a1.w *= scale;
    a2.x *= scale; a2.y *= scale; a2.z *= scale; a2.w *= scale;
    a3.x *= scale; a3.y *= scale; a3.z *= scale; a3.w *= scale;
    cvt8(qa, 0, a0, a1);
    cvt8(qa, 8, a2, a3);
  }

  float m[8], l[8];
  v8f o0 = {}, o1 = {};
#pragma unroll
  for (int r = 0; r < 8; ++r) { m[r] = -1e30f; l[r] = 0.f; }
  const v8f czero = {};
  const int dbase = (lane < 16) ? 0 : 16;   // b_kk run start (contiguous d)

  for (int kb = 0; kb < Lq; kb += 32) {
    // K fragments (B layout): pair dim = contiguous head-dim -> 4 x float4
    v16h b0f, b1f;
#pragma unroll
    for (int t = 0; t < 2; ++t) {
      int key = kb + t * 16 + (lane & 15);
      if (key >= Lq) key = Lq - 1;
      const float* kp = Kp + (size_t)key * DMODEL + h * DHEAD + dbase;
      const float4 x0 = *(const float4*)(kp + 0);
      const float4 x1 = *(const float4*)(kp + 4);
      const float4 x2 = *(const float4*)(kp + 8);
      const float4 x3 = *(const float4*)(kp + 12);
      if (t == 0) { cvt8(b0f, 0, x0, x1); cvt8(b0f, 8, x2, x3); }
      else        { cvt8(b1f, 0, x0, x1); cvt8(b1f, 8, x2, x3); }
    }
    v8f s0 = __builtin_amdgcn_wmma_f32_16x16x32_f16(false, qa, false, b0f, (short)0, czero, false, false);
    v8f s1 = __builtin_amdgcn_wmma_f32_16x16x32_f16(false, qa, false, b1f, (short)0, czero, false, false);

    const bool ok0 = (kb +      (lane & 15)) < Lq;
    const bool ok1 = (kb + 16 + (lane & 15)) < Lq;
#pragma unroll
    for (int r = 0; r < 8; ++r) {
      if (!ok0) s0[r] = -1e30f;
      if (!ok1) s1[r] = -1e30f;
    }
    // online softmax per row (rows live across 16-lane half-groups)
#pragma unroll
    for (int r = 0; r < 8; ++r) {
      float v = fmaxf(s0[r], s1[r]);
      v = fmaxf(v, __shfl_xor(v, 1));
      v = fmaxf(v, __shfl_xor(v, 2));
      v = fmaxf(v, __shfl_xor(v, 4));
      v = fmaxf(v, __shfl_xor(v, 8));
      const float mn = fmaxf(m[r], v);
      const float alpha = __expf(m[r] - mn);
      m[r] = mn;
      l[r] *= alpha;
      o0[r] *= alpha;
      o1[r] *= alpha;
      s0[r] = __expf(s0[r] - mn);
      s1[r] = __expf(s1[r] - mn);
      float rs = s0[r] + s1[r];
      rs += __shfl_xor(rs, 1);
      rs += __shfl_xor(rs, 2);
      rs += __shfl_xor(rs, 4);
      rs += __shfl_xor(rs, 8);
      l[r] += rs;
    }
    // transpose P: C layout -> LDS -> A layout (f16)
    __syncthreads();
    {
      const int rbase = (lane < 16) ? 0 : 8;
      const int c0 = lane & 15;
#pragma unroll
      for (int r = 0; r < 8; ++r) {
        pbuf[rbase + r][c0]      = s0[r];
        pbuf[rbase + r][16 + c0] = s1[r];
      }
    }
    __syncthreads();
    v16h pa;
    {
      const int prow = lane & 15;
#pragma unroll
      for (int i = 0; i < 8; ++i) {
        const int kk = b0 + 2 * i;
        pa[2 * i]     = (_Float16)pbuf[prow][kk];
        pa[2 * i + 1] = (_Float16)pbuf[prow][kk + 1];
      }
    }
    // V fragments (B layout over keys): strided pair loads, clamped keys
    v16h vb0, vb1;
#pragma unroll
    for (int t = 0; t < 2; ++t) {
      const int dcol = t * 16 + (lane & 15);
#pragma unroll
      for (int j = 0; j < 8; ++j) {
        int k0i = kb + b_kk(lane, j);
        int k1i = k0i + 1;
        if (k0i >= Lq) k0i = Lq - 1;
        if (k1i >= Lq) k1i = Lq - 1;
        const _Float16 x0 = (_Float16)Vp[(size_t)k0i * DMODEL + h * DHEAD + dcol];
        const _Float16 x1 = (_Float16)Vp[(size_t)k1i * DMODEL + h * DHEAD + dcol];
        if (t == 0) { vb0[2 * j] = x0; vb0[2 * j + 1] = x1; }
        else        { vb1[2 * j] = x0; vb1[2 * j + 1] = x1; }
      }
    }
    o0 = __builtin_amdgcn_wmma_f32_16x16x32_f16(false, pa, false, vb0, (short)0, o0, false, false);
    o1 = __builtin_amdgcn_wmma_f32_16x16x32_f16(false, pa, false, vb1, (short)0, o1, false, false);
  }

#pragma unroll
  for (int r = 0; r < 8; ++r) {
    const int row = q0 + r + (lane < 16 ? 0 : 8);
    if (row < Lq) {
      const float inv = 1.f / l[r];
      O[(size_t)row * DMODEL + h * DHEAD +      (lane & 15)] = o0[r] * inv;
      O[(size_t)row * DMODEL + h * DHEAD + 16 + (lane & 15)] = o1[r] * inv;
    }
  }
}

// =======================================================================
// Fused residual + LayerNorm: dst = LN(res + x) * s + b ; one wave per row
// =======================================================================
__global__ void res_ln(const float* __restrict__ res, const float* __restrict__ x,
                       const float* __restrict__ s, const float* __restrict__ b,
                       float* __restrict__ dst, int M) {
  const int lane = threadIdx.x & 31;
  const int wid = threadIdx.x >> 5;
  const int row = blockIdx.x * 8 + wid;
  if (row >= M) return;
  const float* rp = res + (size_t)row * DMODEL;
  const float* xp = x + (size_t)row * DMODEL;
  float v[8];
  float sum = 0.f;
#pragma unroll
  for (int i = 0; i < 8; ++i) {
    v[i] = rp[lane + i * 32] + xp[lane + i * 32];
    sum += v[i];
  }
#pragma unroll
  for (int msk = 1; msk < 32; msk <<= 1) sum += __shfl_xor(sum, msk);
  const float mean = sum * (1.f / DMODEL);
  float var = 0.f;
#pragma unroll
  for (int i = 0; i < 8; ++i) { const float d = v[i] - mean; var += d * d; }
#pragma unroll
  for (int msk = 1; msk < 32; msk <<= 1) var += __shfl_xor(var, msk);
  const float inv = rsqrtf(var * (1.f / DMODEL) + 1e-5f);
#pragma unroll
  for (int i = 0; i < 8; ++i) {
    const int c = lane + i * 32;
    dst[(size_t)row * DMODEL + c] = (v[i] - mean) * inv * s[c] + b[c];
  }
}

__global__ void add2(const float* __restrict__ a, const float* __restrict__ b,
                     float* __restrict__ c, int n) {
  const int i = blockIdx.x * blockDim.x + threadIdx.x;
  if (i < n) c[i] = a[i] + b[i];
}

// =======================================================================
// base reference + temporal offsets (per query, level)
// =======================================================================
__global__ void build_base(const float* __restrict__ ref, const float* __restrict__ vr,
                           float* __restrict__ baseb, float* __restrict__ toffb) {
  const int idx = blockIdx.x * blockDim.x + threadIdx.x;  // q*LVL + l
  if (idx >= LQn * LVL) return;
  const int q = idx / LVL, l = idx & (LVL - 1);
  const int t = q / 300;
  const float vrx = vr[l * 2 + 0], vry = vr[l * 2 + 1];
  baseb[idx * 2 + 0] = ref[q * 2 + 0];                                  // + 0*vrx
  baseb[idx * 2 + 1] = ref[q * 2 + 1] * (1.f / TFR) + ((float)t / TFR) * vry;
  const int t2[TWn] = {-2, -1, 1, 2};
  const float first[TWn] = {2.f, 1.f, 1.f, 2.f};
#pragma unroll
  for (int j = 0; j < TWn; ++j) {
    float d;
    if (t == 0)            d = first[j];
    else if (t == TFR - 1) d = -first[j];
    else {
      const int sft = t + t2[j];
      d = (sft < 0) ? (float)(-t) : ((sft > TFR - 1) ? (float)(TFR - 1 - t) : (float)t2[j]);
    }
    toffb[(idx * TWn + j) * 2 + 0] = 0.f * vrx;
    toffb[(idx * TWn + j) * 2 + 1] = (d / TFR) * vry;
  }
}

// =======================================================================
// MSDA: per-(q,head) wave, lane = channel; softmax over 32 (level,point)
// weights computed uniformly; coalesced f16 gathers for bilinear corners.
// =======================================================================
__global__ void msda_sample(const _Float16* __restrict__ V, const float* __restrict__ offp,
                            const float* __restrict__ attp, const float* __restrict__ baseb,
                            const float* __restrict__ toffb, float* __restrict__ Out) {
  const int q = blockIdx.x, h = blockIdx.y, lane = threadIdx.x;
  const int   widthsI[LVL] = {13824, 3456, 864, 216};
  const float widthsF[LVL] = {13824.f, 3456.f, 864.f, 216.f};
  const int   startsI[LVL] = {0, 82944, 103680, 108864};

  // softmax over the 32 attention logits for (q,h) -- uniform across lanes
  const float* ap = attp + (size_t)q * DMODEL + h * 32;
  float aval[32];
  float amax = -1e30f;
#pragma unroll
  for (int j = 0; j < 32; ++j) { aval[j] = ap[j]; amax = fmaxf(amax, aval[j]); }
  float asum = 0.f;
#pragma unroll
  for (int j = 0; j < 32; ++j) { aval[j] = __expf(aval[j] - amax); asum += aval[j]; }
  const float ainv = 1.f / asum;

  float acc = 0.f;
  for (int l = 0; l < LVL; ++l) {
    const float bx = baseb[(q * LVL + l) * 2 + 0];
    const float by = baseb[(q * LVL + l) * 2 + 1];
    const float Wl = widthsF[l];
    const int   Wi = widthsI[l];
    const int   st = startsI[l];
    const float* op = offp + (size_t)q * (NHEAD * LVL * PTn * 2) + (h * LVL + l) * (PTn * 2);
#pragma unroll
    for (int p = 0; p < PTn; ++p) {
      float lx = bx + op[p * 2 + 0] / Wl;
      float ly = by + op[p * 2 + 1] * (1.f / TFR);
      if (p >= NPn) {
        lx += toffb[((q * LVL + l) * TWn + (p - NPn)) * 2 + 0];
        ly += toffb[((q * LVL + l) * TWn + (p - NPn)) * 2 + 1];
      }
      const float x = lx * Wl - 0.5f;
      const float y = ly * TFR - 0.5f;
      const float x0f = floorf(x), y0f = floorf(y);
      const int x0 = (int)x0f, y0 = (int)y0f;
      const float fx = x - x0f, fy = y - y0f;
      float sv = 0.f;
#pragma unroll
      for (int cy = 0; cy < 2; ++cy)
#pragma unroll
        for (int cx = 0; cx < 2; ++cx) {
          const int xi = x0 + cx, yj = y0 + cy;
          if (xi >= 0 && xi < Wi && yj >= 0 && yj < TFR) {
            const float wgt = (cx ? fx : 1.f - fx) * (cy ? fy : 1.f - fy);
            const float vv = (float)V[((size_t)(st + yj * Wi + xi)) * DMODEL + h * DHEAD + lane];
            sv += wgt * vv;
          }
        }
      acc += aval[l * PTn + p] * sv;
    }
  }
  Out[(size_t)q * DMODEL + h * DHEAD + lane] = acc * ainv;
}

// =======================================================================
// host orchestration
// =======================================================================
static void gemm(hipStream_t st, const float* A, const _Float16* Wp, const float* bias,
                 float* Cf, _Float16* Ch, int M, int N, int K, int relu) {
  dim3 g((M + 16 * GMT - 1) / (16 * GMT), N / (16 * GNT));
  hipLaunchKernelGGL(gemm_wmma, g, dim3(32), 0, st, A, Wp, bias, Cf, Ch, M, N, K, relu);
}

static void pack_w(hipStream_t st, const float* W, _Float16* Wp, int K, int N) {
  const int n = K * N;
  hipLaunchKernelGGL(pack_w_f16, dim3((n + 255) / 256), dim3(256), 0, st, W, Wp, K, N);
}

extern "C" void kernel_launch(void* const* d_in, const int* in_sizes, int n_in,
                              void* d_out, int out_size, void* d_ws, size_t ws_size,
                              hipStream_t stream) {
  (void)in_sizes; (void)n_in; (void)out_size; (void)ws_size;
  const float* tgt    = (const float*)d_in[0];
  const float* qpos   = (const float*)d_in[1];
  const float* ref    = (const float*)d_in[2];
  const float* memory = (const float*)d_in[3];
  const float* vr     = (const float*)d_in[4];
  const float* Wq = (const float*)d_in[5];   const float* bq = (const float*)d_in[6];
  const float* Wk = (const float*)d_in[7];   const float* bk = (const float*)d_in[8];
  const float* Wv = (const float*)d_in[9];   const float* bv = (const float*)d_in[10];
  const float* Wo = (const float*)d_in[11];  const float* bo = (const float*)d_in[12];
  const float* Wval = (const float*)d_in[13];const float* bval = (const float*)d_in[14];
  const float* Woff = (const float*)d_in[15];const float* boff = (const float*)d_in[16];
  const float* Watt = (const float*)d_in[17];const float* batt = (const float*)d_in[18];
  const float* Wout = (const float*)d_in[19];const float* bout = (const float*)d_in[20];
  const float* W1 = (const float*)d_in[21];  const float* b1 = (const float*)d_in[22];
  const float* W2 = (const float*)d_in[23];  const float* b2 = (const float*)d_in[24];
  const float* ln1s = (const float*)d_in[25];const float* ln1b = (const float*)d_in[26];
  const float* ln2s = (const float*)d_in[27];const float* ln2b = (const float*)d_in[28];
  const float* ln3s = (const float*)d_in[29];const float* ln3b = (const float*)d_in[30];

  // workspace carve
  char* p = (char*)d_ws;
  auto carve = [&](size_t bytes) -> void* {
    void* r = (void*)p;
    p += (bytes + 255) & ~(size_t)255;
    return r;
  };
  const size_t QD = (size_t)LQn * DMODEL * sizeof(float);
  float* out   = (float*)carve(QD);
  float* qb    = (float*)carve(QD);
  float* Qp    = (float*)carve(QD);
  float* Kp    = (float*)carve(QD);
  float* Vp    = (float*)carve(QD);
  float* attno = (float*)carve(QD);
  float* proj  = (float*)carve(QD);
  float* msdab = (float*)carve(QD);
  float* attp  = (float*)carve(QD);
  float* offp  = (float*)carve((size_t)LQn * 512 * sizeof(float));
  float* h1    = (float*)carve((size_t)LQn * DFF * sizeof(float));
  float* baseb = (float*)carve((size_t)LQn * LVL * 2 * sizeof(float));
  float* toffb = (float*)carve((size_t)LQn * LVL * TWn * 2 * sizeof(float));
  _Float16* vproj = (_Float16*)carve((size_t)LENn * DMODEL * sizeof(_Float16));
  // packed f16 weights, per layer
  const size_t SZ_DD = (size_t)DMODEL * DMODEL;
  _Float16 *pWq[NLAYER], *pWk[NLAYER], *pWv[NLAYER], *pWo[NLAYER], *pWval[NLAYER],
           *pWatt[NLAYER], *pWout[NLAYER], *pWoff[NLAYER], *pW1[NLAYER], *pW2[NLAYER];
  for (int i = 0; i < NLAYER; ++i) {
    pWq[i]   = (_Float16*)carve(SZ_DD * 2);
    pWk[i]   = (_Float16*)carve(SZ_DD * 2);
    pWv[i]   = (_Float16*)carve(SZ_DD * 2);
    pWo[i]   = (_Float16*)carve(SZ_DD * 2);
    pWval[i] = (_Float16*)carve(SZ_DD * 2);
    pWatt[i] = (_Float16*)carve(SZ_DD * 2);
    pWout[i] = (_Float16*)carve(SZ_DD * 2);
    pWoff[i] = (_Float16*)carve((size_t)DMODEL * 512 * 2);
    pW1[i]   = (_Float16*)carve((size_t)DMODEL * DFF * 2);
    pW2[i]   = (_Float16*)carve((size_t)DFF * DMODEL * 2);
  }

  // repack all weights (runs once per call; overlaps nothing but is tiny)
  for (int i = 0; i < NLAYER; ++i) {
    const size_t dd = (size_t)i * SZ_DD;
    pack_w(stream, Wq + dd,   pWq[i],   DMODEL, DMODEL);
    pack_w(stream, Wk + dd,   pWk[i],   DMODEL, DMODEL);
    pack_w(stream, Wv + dd,   pWv[i],   DMODEL, DMODEL);
    pack_w(stream, Wo + dd,   pWo[i],   DMODEL, DMODEL);
    pack_w(stream, Wval + dd, pWval[i], DMODEL, DMODEL);
    pack_w(stream, Watt + dd, pWatt[i], DMODEL, DMODEL);
    pack_w(stream, Wout + dd, pWout[i], DMODEL, DMODEL);
    pack_w(stream, Woff + (size_t)i * DMODEL * 512, pWoff[i], DMODEL, 512);
    pack_w(stream, W1 + (size_t)i * DMODEL * DFF,   pW1[i],   DMODEL, DFF);
    pack_w(stream, W2 + (size_t)i * DFF * DMODEL,   pW2[i],   DFF, DMODEL);
  }

  hipMemcpyAsync(out, tgt, QD, hipMemcpyDeviceToDevice, stream);
  hipLaunchKernelGGL(build_base, dim3((LQn * LVL + 255) / 256), dim3(256), 0, stream,
                     ref, vr, baseb, toffb);

  const int nElem = LQn * DMODEL;
  const dim3 eg((nElem + 255) / 256), eb(256);
  const dim3 lng((LQn + 7) / 8), lnb(256);
  const dim3 fag((LQn + 15) / 16, NHEAD), fab(32);
  const dim3 msg(LQn, NHEAD), msb(32);

  for (int i = 0; i < NLAYER; ++i) {
    const size_t dv = (size_t)i * DMODEL;
    // ---- self attention ----
    hipLaunchKernelGGL(add2, eg, eb, 0, stream, out, qpos, qb, nElem);
    gemm(stream, qb,  pWq[i], bq + dv, Qp, nullptr, LQn, DMODEL, DMODEL, 0);
    gemm(stream, qb,  pWk[i], bk + dv, Kp, nullptr, LQn, DMODEL, DMODEL, 0);
    gemm(stream, out, pWv[i], bv + dv, Vp, nullptr, LQn, DMODEL, DMODEL, 0);
    hipLaunchKernelGGL(flash_attn, fag, fab, 0, stream, Qp, Kp, Vp, attno, LQn);
    gemm(stream, attno, pWo[i], bo + dv, proj, nullptr, LQn, DMODEL, DMODEL, 0);
    hipLaunchKernelGGL(res_ln, lng, lnb, 0, stream, out, proj, ln2s + dv, ln2b + dv, out, LQn);
    // ---- deformable cross attention ----
    hipLaunchKernelGGL(add2, eg, eb, 0, stream, out, qpos, qb, nElem);
    gemm(stream, memory, pWval[i], bval + dv, nullptr, vproj, LENn, DMODEL, DMODEL, 0);
    gemm(stream, qb, pWoff[i], boff + (size_t)i * 512, offp, nullptr, LQn, 512, DMODEL, 0);
    gemm(stream, qb, pWatt[i], batt + dv, attp, nullptr, LQn, DMODEL, DMODEL, 0);
    hipLaunchKernelGGL(msda_sample, msg, msb, 0, stream, vproj, offp, attp, baseb, toffb, msdab);
    gemm(stream, msdab, pWout[i], bout + dv, proj, nullptr, LQn, DMODEL, DMODEL, 0);
    hipLaunchKernelGGL(res_ln, lng, lnb, 0, stream, out, proj, ln1s + dv, ln1b + dv, out, LQn);
    // ---- FFN ----
    gemm(stream, out, pW1[i], b1 + (size_t)i * DFF, h1, nullptr, LQn, DFF, DMODEL, 1);
    gemm(stream, h1, pW2[i], b2 + dv, proj, nullptr, LQn, DMODEL, DFF, 0);
    float* dst = (i == NLAYER - 1) ? (float*)d_out : out;
    hipLaunchKernelGGL(res_ln, lng, lnb, 0, stream, out, proj, ln3s + dv, ln3b + dv, dst, LQn);
  }
}
